// SwinTransformer_25159918420362
// MI455X (gfx1250) — compile-verified, matrix operand order checked by reading, not run on hardware
//
#include <hip/hip_runtime.h>

typedef __attribute__((ext_vector_type(16))) __bf16 v16bf;
typedef __attribute__((ext_vector_type(8)))  float  v8f;

namespace swin {

constexpr int   kDim    = 256;
constexpr int   kHeads  = 8;
constexpr int   kHd     = 32;
constexpr int   kNtok   = 64;       // tokens per window
constexpr int   kBatch  = 16;
constexpr int   kNwin   = kBatch * 64;     // 1024 windows total
constexpr int   kMRows  = kNwin * kNtok;   // 65536 rows
constexpr int   kMlp    = 1024;
constexpr int   kShift  = 4;
constexpr float kScale  = 0.17677669529663687f;   // 32^-0.5

// ---------------- workspace layout (bytes) ----------------
constexpr size_t kWblkElems = 768*256 + 256*256 + 1024*256 + 256*1024; // 786432
constexpr size_t OFF_W    = 0;                                     // bf16 weights (both blocks)
constexpr size_t OFF_BIAS = OFF_W + 2ull * kWblkElems * 2;         // f32 bias tables
constexpr size_t OFF_X    = OFF_BIAS + 2ull * kHeads * 64 * 64 * 4;
constexpr size_t OFF_A    = OFF_X + (size_t)kMRows * kDim * 4;     // bf16 activation buffer
constexpr size_t OFF_BIG  = OFF_A + (size_t)kMRows * kDim * 2;     // q|k|vT  (later: mlp hidden)
constexpr size_t QK_ELEMS = (size_t)kNwin * kHeads * kNtok * kHd;  // 16,777,216

// ---------------- fragment helpers ----------------
// A/B fragment (16-bit, 16x32): lane = (row or col) & 15 ; lane>>4 selects K-group.
// elements 0..7  = K[k0 .. k0+7], elements 8..15 = K[k0+16 .. k0+23]
__device__ inline v16bf load_frag(const __bf16* p) {
  union { uint4 u[2]; v16bf v; } f;
  f.u[0] = *reinterpret_cast<const uint4*>(p);
  f.u[1] = *reinterpret_cast<const uint4*>(p + 16);
  return f.v;
}

__device__ inline v8f zero8() {
  v8f z;
#pragma unroll
  for (int i = 0; i < 8; ++i) z[i] = 0.f;
  return z;
}

__device__ inline v8f wmma_bf16(v16bf a, v16bf b, v8f c) {
  return __builtin_amdgcn_wmma_f32_16x16x32_bf16(false, a, false, b, (short)0, c,
                                                 false, false);
}

// One wave computes a 64x64 f32 tile of A[M,K] @ Bt[N,K]^T
template <int KDIM>
__device__ inline void gemm64x64(const __bf16* __restrict__ A,
                                 const __bf16* __restrict__ Bt,
                                 int rowBase, int colBase, v8f acc[4][4]) {
  const int lane = threadIdx.x & 31;
  const int lr = lane & 15;
  const int lh = lane >> 4;
  for (int k0 = 0; k0 < KDIM; k0 += 32) {
    v16bf a[4], b[4];
#pragma unroll
    for (int t = 0; t < 4; ++t)
      a[t] = load_frag(A + (size_t)(rowBase + t*16 + lr) * KDIM + k0 + lh*8);
#pragma unroll
    for (int t = 0; t < 4; ++t)
      b[t] = load_frag(Bt + (size_t)(colBase + t*16 + lr) * KDIM + k0 + lh*8);
#pragma unroll
    for (int mt = 0; mt < 4; ++mt)
#pragma unroll
      for (int nt = 0; nt < 4; ++nt)
        acc[mt][nt] = wmma_bf16(a[mt], b[nt], acc[mt][nt]);
  }
}

// ---------------- kernels ----------------

// transpose + f32->bf16:  Wt[o*K + k] = W[k*N + o]
__global__ void k_wt(const float* __restrict__ W, __bf16* __restrict__ Wt,
                     int K, int N) {
  int idx = blockIdx.x * 256 + threadIdx.x;
  if (idx >= K * N) return;
  int o = idx / K, kk = idx - o * K;
  Wt[(size_t)o * K + kk] = (__bf16)W[(size_t)kk * N + o];
}

// relative position bias table:  tab[head][n][m] = rpb[relidx(n,m)][head]
__global__ void k_bias(const float* __restrict__ rpb, float* __restrict__ tab) {
  int idx = blockIdx.x * 256 + threadIdx.x;          // 8*64*64 = 32768
  int head = idx >> 12;
  int rem = idx & 4095;
  int n = rem >> 6, m = rem & 63;
  int ri = ((n >> 3) - (m >> 3) + 7) * 15 + ((n & 7) - (m & 7) + 7);
  tab[idx] = rpb[ri * kHeads + head];
}

// LayerNorm (+ optional cyclic shift & window partition): one wave per token.
__global__ void __launch_bounds__(128)
k_ln(const float* __restrict__ x, const float* __restrict__ g,
     const float* __restrict__ bt, __bf16* __restrict__ out,
     int shift, int windowed) {
  int wave = (blockIdx.x * blockDim.x + threadIdx.x) >> 5;
  int lane = threadIdx.x & 31;
  size_t srcTok;
  if (windowed) {
    int win = wave >> 6, n = wave & 63;
    int bi = win >> 6, wi = win & 63;
    int hh = (wi >> 3) * 8 + (n >> 3);
    int ww = (wi & 7) * 8 + (n & 7);
    int h = (hh + shift) & 63, w = (ww + shift) & 63;   // roll(x, -shift)
    srcTok = (size_t)bi * 4096 + h * 64 + w;
  } else {
    srcTok = (size_t)wave;
  }
  const float* src = x + srcTok * kDim;
  float v[8];
  float s = 0.f, s2 = 0.f;
#pragma unroll
  for (int i = 0; i < 8; ++i) {
    v[i] = src[lane * 8 + i];
    s += v[i]; s2 += v[i] * v[i];
  }
  for (int off = 16; off; off >>= 1) {
    s  += __shfl_xor(s,  off, 32);
    s2 += __shfl_xor(s2, off, 32);
  }
  float mu = s * (1.f / kDim);
  float var = s2 * (1.f / kDim) - mu * mu;
  float rstd = rsqrtf(var + 1e-5f);
  __bf16* dst = out + (size_t)wave * kDim;
#pragma unroll
  for (int i = 0; i < 8; ++i) {
    int c = lane * 8 + i;
    dst[c] = (__bf16)((v[i] - mu) * rstd * g[c] + bt[c]);
  }
}

// QKV GEMM: A[65536,256] @ Wt[768,256]^T + bias; split q(*scale)/k/vT layouts
__global__ void __launch_bounds__(128)
k_gemm_qkv(const __bf16* __restrict__ A, const __bf16* __restrict__ Bt,
           const float* __restrict__ bias, __bf16* __restrict__ q,
           __bf16* __restrict__ k, __bf16* __restrict__ vT) {
  int gw = blockIdx.x * 4 + (threadIdx.x >> 5);
  const int CT = 768 / 64;
  int rowBase = (gw / CT) * 64, colBase = (gw % CT) * 64;
  v8f acc[4][4];
#pragma unroll
  for (int a = 0; a < 4; ++a)
#pragma unroll
    for (int b = 0; b < 4; ++b) acc[a][b] = zero8();
  gemm64x64<256>(A, Bt, rowBase, colBase, acc);
  int lane = threadIdx.x & 31, lr = lane & 15, lh = lane >> 4;
#pragma unroll
  for (int mt = 0; mt < 4; ++mt)
#pragma unroll
    for (int nt = 0; nt < 4; ++nt)
#pragma unroll
      for (int j = 0; j < 8; ++j) {
        int row = rowBase + mt * 16 + j + lh * 8;
        int col = colBase + nt * 16 + lr;
        float val = acc[mt][nt][j] + bias[col];
        int s = col >> 8, hc = col & 255, head = hc >> 5, hd = hc & 31;
        int win = row >> 6, n = row & 63;
        size_t bh = (size_t)(win * kHeads + head);
        if (s == 0)      q [(bh * 64 + n) * 32 + hd] = (__bf16)(val * kScale);
        else if (s == 1) k [(bh * 64 + n) * 32 + hd] = (__bf16)val;
        else             vT[(bh * 32 + hd) * 64 + n] = (__bf16)val;
      }
}

// Attention: one wave per (window, head). QK^T -> bias(+mask) -> softmax -> PV.
__global__ void __launch_bounds__(64)
k_attn(const __bf16* __restrict__ q, const __bf16* __restrict__ kk,
       const __bf16* __restrict__ vT, const float* __restrict__ biasTab,
       __bf16* __restrict__ ao, int shifted) {
  __shared__ char smem[2 * 24576];                 // per wave: 16KB f32 S + 8KB bf16 P
  int widx = threadIdx.x >> 5;
  int lane = threadIdx.x & 31;
  int lr = lane & 15, lh = lane >> 4;
  int gw = blockIdx.x * 2 + widx;
  int win = gw >> 3, head = gw & 7;
  float*  Sf = reinterpret_cast<float*>(smem + widx * 24576);
  __bf16* Pb = reinterpret_cast<__bf16*>(smem + widx * 24576 + 16384);
  size_t bh = (size_t)(win * kHeads + head);
  const __bf16* qp = q  + bh * 64 * 32;
  const __bf16* kp = kk + bh * 64 * 32;
  const __bf16* vp = vT + bh * 32 * 64;

  v16bf aq[4], bk[4];
#pragma unroll
  for (int t = 0; t < 4; ++t) aq[t] = load_frag(qp + (size_t)(t*16 + lr) * 32 + lh*8);
#pragma unroll
  for (int t = 0; t < 4; ++t) bk[t] = load_frag(kp + (size_t)(t*16 + lr) * 32 + lh*8);

  int wi = win & 63;
  int wr = (wi >> 3) * 8, wc = (wi & 7) * 8;
  const float* bTab = biasTab + head * 4096;

#pragma unroll
  for (int mt = 0; mt < 4; ++mt)
#pragma unroll
    for (int nt = 0; nt < 4; ++nt) {
      v8f s = wmma_bf16(aq[mt], bk[nt], zero8());
#pragma unroll
      for (int j = 0; j < 8; ++j) {
        int n = mt * 16 + j + lh * 8;
        int m = nt * 16 + lr;
        float val = s[j] + bTab[n * 64 + m];
        if (shifted) {
          int hn = wr + (n >> 3), wn = wc + (n & 7);
          int hm = wr + (m >> 3), wm = wc + (m & 7);
          int ln = ((hn < 56) ? 0 : (hn < 60) ? 1 : 2) * 3 +
                   ((wn < 56) ? 0 : (wn < 60) ? 1 : 2);
          int lm = ((hm < 56) ? 0 : (hm < 60) ? 1 : 2) * 3 +
                   ((wm < 56) ? 0 : (wm < 60) ? 1 : 2);
          if (ln != lm) val -= 100.f;
        }
        Sf[n * 64 + m] = val;
      }
    }
  __syncthreads();

  // softmax: each lane owns 2 rows
#pragma unroll
  for (int rr = 0; rr < 2; ++rr) {
    int r = lane * 2 + rr;
    float mx = -3.0e38f;
    for (int i = 0; i < 64; ++i) mx = fmaxf(mx, Sf[r * 64 + i]);
    float sum = 0.f;
    for (int i = 0; i < 64; ++i) sum += __expf(Sf[r * 64 + i] - mx);
    float inv = 1.f / sum;
    for (int i = 0; i < 64; ++i)
      Pb[r * 64 + i] = (__bf16)(__expf(Sf[r * 64 + i] - mx) * inv);
  }
  __syncthreads();

  // O = P(64x64) @ V(64x32)
  v8f o[4][2];
#pragma unroll
  for (int a = 0; a < 4; ++a)
#pragma unroll
    for (int b = 0; b < 2; ++b) o[a][b] = zero8();
#pragma unroll
  for (int ks = 0; ks < 2; ++ks) {
    v16bf ap[4], bv[2];
#pragma unroll
    for (int t = 0; t < 4; ++t)
      ap[t] = load_frag(Pb + (size_t)(t*16 + lr) * 64 + ks*32 + lh*8);
#pragma unroll
    for (int t = 0; t < 2; ++t)
      bv[t] = load_frag(vp + (size_t)(t*16 + lr) * 64 + ks*32 + lh*8);
#pragma unroll
    for (int mt = 0; mt < 4; ++mt)
#pragma unroll
      for (int nt = 0; nt < 2; ++nt) o[mt][nt] = wmma_bf16(ap[mt], bv[nt], o[mt][nt]);
  }
#pragma unroll
  for (int mt = 0; mt < 4; ++mt)
#pragma unroll
    for (int nt = 0; nt < 2; ++nt)
#pragma unroll
      for (int j = 0; j < 8; ++j) {
        int n = mt * 16 + j + lh * 8;
        int hd = nt * 16 + lr;
        ao[(size_t)(win * 64 + n) * kDim + head * kHd + hd] = (__bf16)(o[mt][nt][j]);
      }
}

// Proj GEMM + window reverse + un-shift + residual (f32)
__global__ void __launch_bounds__(128)
k_gemm_proj(const __bf16* __restrict__ A, const __bf16* __restrict__ Bt,
            const float* __restrict__ bias, const float* __restrict__ xin,
            float* __restrict__ xout, int shift) {
  int gw = blockIdx.x * 4 + (threadIdx.x >> 5);
  const int CT = 4;
  int rowBase = (gw / CT) * 64, colBase = (gw % CT) * 64;
  v8f acc[4][4];
#pragma unroll
  for (int a = 0; a < 4; ++a)
#pragma unroll
    for (int b = 0; b < 4; ++b) acc[a][b] = zero8();
  gemm64x64<256>(A, Bt, rowBase, colBase, acc);
  int lane = threadIdx.x & 31, lr = lane & 15, lh = lane >> 4;
#pragma unroll
  for (int mt = 0; mt < 4; ++mt)
#pragma unroll
    for (int nt = 0; nt < 4; ++nt)
#pragma unroll
      for (int j = 0; j < 8; ++j) {
        int row = rowBase + mt * 16 + j + lh * 8;
        int col = colBase + nt * 16 + lr;
        float val = acc[mt][nt][j] + bias[col];
        int win = row >> 6, n = row & 63;
        int bi = win >> 6, wi = win & 63;
        int hh = (wi >> 3) * 8 + (n >> 3);
        int ww = (wi & 7) * 8 + (n & 7);
        int h = (hh + shift) & 63, w = (ww + shift) & 63;   // roll back (+shift)
        size_t t = (size_t)bi * 4096 + h * 64 + w;
        xout[t * kDim + col] = xin[t * kDim + col] + val;
      }
}

// FC1 GEMM + exact GELU -> bf16 hidden
__global__ void __launch_bounds__(128)
k_gemm_fc1(const __bf16* __restrict__ A, const __bf16* __restrict__ Bt,
           const float* __restrict__ bias, __bf16* __restrict__ out) {
  int gw = blockIdx.x * 4 + (threadIdx.x >> 5);
  const int CT = kMlp / 64;
  int rowBase = (gw / CT) * 64, colBase = (gw % CT) * 64;
  v8f acc[4][4];
#pragma unroll
  for (int a = 0; a < 4; ++a)
#pragma unroll
    for (int b = 0; b < 4; ++b) acc[a][b] = zero8();
  gemm64x64<256>(A, Bt, rowBase, colBase, acc);
  int lane = threadIdx.x & 31, lr = lane & 15, lh = lane >> 4;
#pragma unroll
  for (int mt = 0; mt < 4; ++mt)
#pragma unroll
    for (int nt = 0; nt < 4; ++nt)
#pragma unroll
      for (int j = 0; j < 8; ++j) {
        int row = rowBase + mt * 16 + j + lh * 8;
        int col = colBase + nt * 16 + lr;
        float x = acc[mt][nt][j] + bias[col];
        float g = 0.5f * x * (1.f + erff(x * 0.70710678118654752f));
        out[(size_t)row * kMlp + col] = (__bf16)g;
      }
}

// FC2 GEMM + residual -> f32 (block output)
__global__ void __launch_bounds__(128)
k_gemm_fc2(const __bf16* __restrict__ A, const __bf16* __restrict__ Bt,
           const float* __restrict__ bias, const float* __restrict__ xmid,
           float* __restrict__ xout) {
  int gw = blockIdx.x * 4 + (threadIdx.x >> 5);
  const int CT = 4;
  int rowBase = (gw / CT) * 64, colBase = (gw % CT) * 64;
  v8f acc[4][4];
#pragma unroll
  for (int a = 0; a < 4; ++a)
#pragma unroll
    for (int b = 0; b < 4; ++b) acc[a][b] = zero8();
  gemm64x64<1024>(A, Bt, rowBase, colBase, acc);
  int lane = threadIdx.x & 31, lr = lane & 15, lh = lane >> 4;
#pragma unroll
  for (int mt = 0; mt < 4; ++mt)
#pragma unroll
    for (int nt = 0; nt < 4; ++nt)
#pragma unroll
      for (int j = 0; j < 8; ++j) {
        int row = rowBase + mt * 16 + j + lh * 8;
        int col = colBase + nt * 16 + lr;
        float val = acc[mt][nt][j] + bias[col];
        xout[(size_t)row * kDim + col] = xmid[(size_t)row * kDim + col] + val;
      }
}

}  // namespace swin

extern "C" void kernel_launch(void* const* d_in, const int* in_sizes, int n_in,
                              void* d_out, int out_size, void* d_ws, size_t ws_size,
                              hipStream_t stream) {
  using namespace swin;
  (void)in_sizes; (void)n_in; (void)out_size; (void)ws_size;

  char* ws = (char*)d_ws;
  __bf16* wbase   = (__bf16*)(ws + OFF_W);
  float*  biasTab = (float*)(ws + OFF_BIAS);
  float*  xbuf    = (float*)(ws + OFF_X);
  __bf16* Abuf    = (__bf16*)(ws + OFF_A);
  __bf16* qbuf    = (__bf16*)(ws + OFF_BIG);
  __bf16* kbuf    = qbuf + QK_ELEMS;
  __bf16* vTbuf   = kbuf + QK_ELEMS;
  __bf16* hmid    = (__bf16*)(ws + OFF_BIG);   // reuses q/k/vT region (dead by then)

  for (int blk = 0; blk < 2; ++blk) {
    const int pi = 1 + blk * 13;
    const float* n1g  = (const float*)d_in[pi + 0];
    const float* n1b  = (const float*)d_in[pi + 1];
    const float* qkvw = (const float*)d_in[pi + 2];
    const float* qkvb = (const float*)d_in[pi + 3];
    const float* rpb  = (const float*)d_in[pi + 4];
    const float* pw   = (const float*)d_in[pi + 5];
    const float* pb   = (const float*)d_in[pi + 6];
    const float* n2g  = (const float*)d_in[pi + 7];
    const float* n2b  = (const float*)d_in[pi + 8];
    const float* f1w  = (const float*)d_in[pi + 9];
    const float* f1b  = (const float*)d_in[pi + 10];
    const float* f2w  = (const float*)d_in[pi + 11];
    const float* f2b  = (const float*)d_in[pi + 12];

    __bf16* qkvwT = wbase + (size_t)blk * kWblkElems;
    __bf16* pwT   = qkvwT + 768 * 256;
    __bf16* f1wT  = pwT + 256 * 256;
    __bf16* f2wT  = f1wT + 1024 * 256;
    float*  bTab  = biasTab + (size_t)blk * kHeads * 64 * 64;

    const int shift = blk ? kShift : 0;
    const float* xin = (blk == 0) ? (const float*)d_in[0] : (const float*)xbuf;
    float* xfin = (blk == 0) ? xbuf : (float*)d_out;

    k_wt<<<(768 * 256) / 256, 256, 0, stream>>>(qkvw, qkvwT, 256, 768);
    k_wt<<<(256 * 256) / 256, 256, 0, stream>>>(pw, pwT, 256, 256);
    k_wt<<<(256 * 1024) / 256, 256, 0, stream>>>(f1w, f1wT, 256, 1024);
    k_wt<<<(1024 * 256) / 256, 256, 0, stream>>>(f2w, f2wT, 1024, 256);
    k_bias<<<(kHeads * 64 * 64) / 256, 256, 0, stream>>>(rpb, bTab);

    // LN1 + shift + window partition -> Abuf (bf16)
    k_ln<<<kMRows / 4, 128, 0, stream>>>(xin, n1g, n1b, Abuf, shift, 1);
    // QKV
    k_gemm_qkv<<<(kMRows / 64) * (768 / 64) / 4, 128, 0, stream>>>(
        Abuf, qkvwT, qkvb, qbuf, kbuf, vTbuf);
    // Attention -> Abuf (bf16, reused as attention-output buffer)
    k_attn<<<(kNwin * kHeads) / 2, 64, 0, stream>>>(qbuf, kbuf, vTbuf, bTab, Abuf,
                                                    blk);
    // Proj + window reverse + residual -> xbuf (f32)
    k_gemm_proj<<<(kMRows / 64) * (256 / 64) / 4, 128, 0, stream>>>(
        Abuf, pwT, pb, xin, xbuf, shift);
    // LN2 -> Abuf (bf16)
    k_ln<<<kMRows / 4, 128, 0, stream>>>(xbuf, n2g, n2b, Abuf, 0, 0);
    // FC1 + GELU -> hmid (bf16)
    k_gemm_fc1<<<(kMRows / 64) * (kMlp / 64) / 4, 128, 0, stream>>>(Abuf, f1wT,
                                                                    f1b, hmid);
    // FC2 + residual -> xbuf (blk0) or d_out (blk1)
    k_gemm_fc2<<<(kMRows / 64) * (256 / 64) / 4, 128, 0, stream>>>(hmid, f2wT,
                                                                   f2b, xbuf, xfin);
  }
}
